// ContrastiveClusteringLoss_7507602833891
// MI455X (gfx1250) — matrix operand order repaired and checked by reading, not compiled, not run on hardware
//
#include <hip/hip_runtime.h>
#include <hip/hip_bf16.h>
#include <math.h>

typedef _Float16 v16h __attribute__((ext_vector_type(16)));
typedef _Float16 h8   __attribute__((ext_vector_type(8)));
typedef float    v8f  __attribute__((ext_vector_type(8)));

#define NROWS 8192
#define DIM   512
#define TILE  128
#define KC    64
#define LDH   72   // halfs per LDS row (144 B: 16B-aligned, bank-staggered)

// CDNA5 async global->LDS DMA (ASYNCcnt path); saddr form:
//   vdst = LDS byte address (VGPR), vaddr = 32-bit byte offset, saddr = 64-bit base
__device__ __forceinline__ void async_ld128(unsigned lds_addr,
                                            const void* base,
                                            unsigned voff) {
    asm volatile("global_load_async_to_lds_b128 %0, %1, %2"
                 :: "v"(lds_addr), "v"(voff), "s"(base)
                 : "memory");
}
__device__ __forceinline__ void wait_asynccnt0() {
    asm volatile("s_wait_asynccnt 0x0" ::: "memory");
}

// ------------------------------------------------------------------
// Kernel 1: zero the accumulators and the output scalar
// ------------------------------------------------------------------
__global__ void ccl_init_kernel(float* __restrict__ pos,
                                float* __restrict__ tot,
                                float* __restrict__ out) {
    int i = blockIdx.x * blockDim.x + threadIdx.x;
    if (i < NROWS) { pos[i] = 0.0f; tot[i] = 0.0f; }
    if (i == 0) out[0] = 0.0f;
}

// ------------------------------------------------------------------
// Kernel 2: L2-normalize each row, store as f16
// ------------------------------------------------------------------
__global__ __launch_bounds__(256)
void ccl_normalize_kernel(const float* __restrict__ X,
                          _Float16* __restrict__ Xh) {
    __shared__ float red[8];
    const int row = blockIdx.x;
    const int t   = threadIdx.x;
    const float* xr = X + (size_t)row * DIM;

    float a = xr[t];
    float b = xr[t + 256];
    float ss = a * a + b * b;
    #pragma unroll
    for (int m = 16; m >= 1; m >>= 1) ss += __shfl_xor(ss, m, 32);
    if ((t & 31) == 0) red[t >> 5] = ss;
    __syncthreads();
    if (t == 0) {
        float s = 0.0f;
        #pragma unroll
        for (int k = 0; k < 8; ++k) s += red[k];
        red[0] = 1.0f / fmaxf(sqrtf(s), 1e-12f);
    }
    __syncthreads();
    const float scale = red[0];
    Xh[(size_t)row * DIM + t]       = (_Float16)(a * scale);
    Xh[(size_t)row * DIM + t + 256] = (_Float16)(b * scale);
}

// ------------------------------------------------------------------
// Kernel 3: fused Gram-matrix (WMMA f16->f32) + exp + masked row sums
// Block = 256 threads (8 waves), 128x128 macro-tile, double-buffered
// LDS fed by global_load_async_to_lds_b128.
// ------------------------------------------------------------------
__global__ __launch_bounds__(256)
void ccl_gram_kernel(const _Float16* __restrict__ Xh,
                     const int*      __restrict__ cl,
                     float* __restrict__ pos,
                     float* __restrict__ tot) {
    __shared__ _Float16 As[2][TILE * LDH];
    __shared__ _Float16 Bs[2][TILE * LDH];

    const int tid     = threadIdx.x;
    const int lane    = tid & 31;
    const int wave    = tid >> 5;
    const int halfSel = (lane >> 4) & 1;
    const int ln      = lane & 15;
    const int iBase   = blockIdx.y * TILE;
    const int jBase   = blockIdx.x * TILE;

    const _Float16* baseA = Xh + (size_t)iBase * DIM;   // uniform -> SGPRs
    const _Float16* baseB = Xh + (size_t)jBase * DIM;

    // per-thread staging pattern: 4 x b128 per tile per chunk
    unsigned goff[4], lAoff[4][2], lBoff[4][2];
    #pragma unroll
    for (int q = 0; q < 4; ++q) {
        int u   = tid + q * 256;          // 0..1023
        int row = u >> 3;                 // 0..127
        int c16 = u & 7;                  // 8 halfs per b128
        goff[q] = (unsigned)(row * (DIM * 2) + c16 * 16);
        #pragma unroll
        for (int b = 0; b < 2; ++b) {
            lAoff[q][b] = (unsigned)(size_t)&As[b][row * LDH + c16 * 8];
            lBoff[q][b] = (unsigned)(size_t)&Bs[b][row * LDH + c16 * 8];
        }
    }

    v8f acc[8] = {};

    // prime buffer 0
    #pragma unroll
    for (int q = 0; q < 4; ++q) {
        async_ld128(lAoff[q][0], baseA, goff[q]);
        async_ld128(lBoff[q][0], baseB, goff[q]);
    }
    wait_asynccnt0();
    __syncthreads();

    for (int k0 = 0; k0 < DIM; k0 += KC) {
        const int cur = (k0 / KC) & 1;
        if (k0 + KC < DIM) {              // stage next chunk into other buffer
            const unsigned kb = (unsigned)((k0 + KC) * 2);
            #pragma unroll
            for (int q = 0; q < 4; ++q) {
                async_ld128(lAoff[q][cur ^ 1], baseA, goff[q] + kb);
                async_ld128(lBoff[q][cur ^ 1], baseB, goff[q] + kb);
            }
        }

        #pragma unroll
        for (int kc = 0; kc < KC; kc += 32) {
            v16h aF;
            h8* ap = (h8*)&aF;
            const _Float16* arow = &As[cur][(wave * 16 + ln) * LDH + kc + halfSel * 8];
            ap[0] = *(const h8*)(arow);        // K = kc + {0..7} / {8..15}
            ap[1] = *(const h8*)(arow + 16);   // K = kc + {16..23} / {24..31}

            #pragma unroll
            for (int t = 0; t < 8; ++t) {
                v16h bF;
                h8* bp = (h8*)&bF;
                const _Float16* brow = &Bs[cur][(t * 16 + ln) * LDH + kc + halfSel * 8];
                bp[0] = *(const h8*)(brow);
                bp[1] = *(const h8*)(brow + 16);
                acc[t] = __builtin_amdgcn_wmma_f32_16x16x32_f16(
                    false, aF, false, bF, (short)0, acc[t], false, false);
            }
        }

        wait_asynccnt0();                  // next buffer landed (this wave)
        __syncthreads();                   // ... and for all waves
    }

    // ---- epilogue: exp(sim/T), masked per-row sums ----
    const int iRow = iBase + wave * 16;
    int ci[8];
    #pragma unroll
    for (int r = 0; r < 8; ++r) ci[r] = cl[iRow + r + 8 * halfSel];

    float tp[8] = {}, pp[8] = {};
    #pragma unroll
    for (int t = 0; t < 8; ++t) {
        const int cj = cl[jBase + t * 16 + ln];
        #pragma unroll
        for (int r = 0; r < 8; ++r) {
            float v = __expf(10.0f * acc[t][r]);   // 1/TEMPERATURE = 10
            tp[r] += v;
            pp[r] += (cj == ci[r]) ? v : 0.0f;
        }
    }
    #pragma unroll
    for (int r = 0; r < 8; ++r) {
        #pragma unroll
        for (int m = 8; m >= 1; m >>= 1) {
            tp[r] += __shfl_xor(tp[r], m, 32);
            pp[r] += __shfl_xor(pp[r], m, 32);
        }
    }
    if (ln == 0) {
        #pragma unroll
        for (int r = 0; r < 8; ++r) {
            int gi = iRow + r + 8 * halfSel;
            atomicAdd(&tot[gi], tp[r]);
            atomicAdd(&pos[gi], pp[r]);
        }
    }
}

// ------------------------------------------------------------------
// Kernel 4: loss_i = -log(pos/(tot+eps)); sum into out[0]
// ------------------------------------------------------------------
__global__ __launch_bounds__(256)
void ccl_finalize_kernel(const float* __restrict__ pos,
                         const float* __restrict__ tot,
                         float* __restrict__ out) {
    __shared__ float red[8];
    int i = blockIdx.x * 256 + threadIdx.x;
    float l = 0.0f;
    if (i < NROWS) {
        float p = pos[i];
        float d = tot[i] + 1e-8f;     // pos + neg + EPS_LOSS
        l = -logf(p / d);
    }
    #pragma unroll
    for (int m = 16; m >= 1; m >>= 1) l += __shfl_xor(l, m, 32);
    if ((threadIdx.x & 31) == 0) red[threadIdx.x >> 5] = l;
    __syncthreads();
    if (threadIdx.x == 0) {
        float s = 0.0f;
        #pragma unroll
        for (int k = 0; k < 8; ++k) s += red[k];
        atomicAdd(out, s);
    }
}

// ------------------------------------------------------------------
extern "C" void kernel_launch(void* const* d_in, const int* in_sizes, int n_in,
                              void* d_out, int out_size, void* d_ws, size_t ws_size,
                              hipStream_t stream) {
    const float* X  = (const float*)d_in[0];   // [8192, 512] f32
    const int*   cl = (const int*)d_in[1];     // [8192] cluster ids
    float* out = (float*)d_out;

    _Float16* Xh  = (_Float16*)d_ws;                               // 8 MiB
    float*    pos = (float*)((char*)d_ws + (size_t)NROWS * DIM * 2);
    float*    tot = pos + NROWS;

    ccl_init_kernel<<<(NROWS + 255) / 256, 256, 0, stream>>>(pos, tot, out);
    ccl_normalize_kernel<<<NROWS, 256, 0, stream>>>(X, Xh);
    dim3 grid(NROWS / TILE, NROWS / TILE);     // 64 x 64 blocks
    ccl_gram_kernel<<<grid, 256, 0, stream>>>(Xh, cl, pos, tot);
    ccl_finalize_kernel<<<NROWS / 256, 256, 0, stream>>>(pos, tot, out);
}
